// Diffusion_65335042506822
// MI455X (gfx1250) — compile-verified, matrix operand order checked by reading, not run on hardware
//
#include <hip/hip_runtime.h>
#include <math.h>

typedef __attribute__((ext_vector_type(16))) _Float16 v16h;
typedef __attribute__((ext_vector_type(8)))  _Float16 v8h;
typedef __attribute__((ext_vector_type(8)))  float    v8f;

#define NRES   1024
#define KNB    48
#define NA     5
#define LOCALD 512
#define PAIRD  128
#define FREAL  514
#define FPAD   544   // 17 * 32
#define EDG    32    // edges per block in k_edges

// Async global->LDS copy via inline asm (bypasses builtin-signature differences
// between toolchains; see CDNA5 ISA and CDNA5 ISA).
__device__ __forceinline__ void async_copy_b128(const float* src, float* dst_lds) {
#if defined(__gfx1250__)
    typedef __attribute__((address_space(3))) unsigned char* LDSP;
    unsigned lds_addr = (unsigned)(unsigned long long)(LDSP)(void*)dst_lds;
    asm volatile("global_load_async_to_lds_b128 %0, %1, off"
                 :: "v"(lds_addr), "v"(src)
                 : "memory");
#else
    *(float4*)dst_lds = *(const float4*)src;
#endif
}

__device__ __forceinline__ void wait_async_all() {
#if defined(__gfx1250__)
    asm volatile("s_wait_asynccnt 0x0" ::: "memory");
#endif
}

__device__ __forceinline__ v16h load_frag(const _Float16* p) {
    // 8 contiguous halves at p (K = base..base+7), 8 at p+16 (K = 16+base..)
    v8h lo = *(const v8h*)p;
    v8h hi = *(const v8h*)(p + 16);
    v16h r;
#pragma unroll
    for (int i = 0; i < 8; ++i) { r[i] = lo[i]; r[i + 8] = hi[i]; }
    return r;
}

__device__ __forceinline__ v8f wmma16(v16h a, v16h b, v8f c) {
    return __builtin_amdgcn_wmma_f32_16x16x32_f16(false, a, false, b,
                                                  (short)0, c, false, false);
}

// ---------------------------------------------------------------------------
// Weight transpose + f32->f16 conversion (N-major, K contiguous).
// ---------------------------------------------------------------------------
__global__ void k_prep_w(const float* __restrict__ Wdist, const float* __restrict__ Wdir,
                         const float* __restrict__ Wrot,  const float* __restrict__ Wvec,
                         const float* __restrict__ w1,    const float* __restrict__ w2,
                         const float* __restrict__ Wli,   const float* __restrict__ Wlj,
                         unsigned short* __restrict__ WtAllU, unsigned short* __restrict__ Wt1U,
                         unsigned short* __restrict__ Wt2U,   unsigned short* __restrict__ WtLiU,
                         unsigned short* __restrict__ WtLjU) {
    _Float16* WtAll = (_Float16*)WtAllU;
    _Float16* Wt1   = (_Float16*)Wt1U;
    _Float16* Wt2   = (_Float16*)Wt2U;
    _Float16* WtLi  = (_Float16*)WtLiU;
    _Float16* WtLj  = (_Float16*)WtLjU;
    int idx = blockIdx.x * 256 + threadIdx.x;

    if (idx < PAIRD * FPAD) {
        int row = idx / FPAD, f = idx % FPAD;
        float v;
        if      (f < 400) v = Wdist[f * PAIRD + row];
        else if (f < 475) v = Wdir[(f - 400) * PAIRD + row];
        else if (f < 484) v = Wrot[(f - 475) * PAIRD + row];
        else if (f < 514) v = Wvec[(f - 484) * PAIRD + row];
        else              v = 0.0f;
        WtAll[idx] = (_Float16)v;
        return;
    }
    idx -= PAIRD * FPAD;
    if (idx < 256 * 128) { int n = idx / 128, k = idx % 128; Wt1[idx]  = (_Float16)w1[k * 256 + n]; return; }
    idx -= 256 * 128;
    if (idx < 128 * 256) { int n = idx / 256, k = idx % 256; Wt2[idx]  = (_Float16)w2[k * 128 + n]; return; }
    idx -= 128 * 256;
    if (idx < 128 * 512) { int n = idx / 512, k = idx % 512; WtLi[idx] = (_Float16)Wli[k * 128 + n]; return; }
    idx -= 128 * 512;
    if (idx < 128 * 512) { int n = idx / 512, k = idx % 512; WtLj[idx] = (_Float16)Wlj[k * 128 + n]; return; }
}

// ---------------------------------------------------------------------------
// Per-residue frames: R (columns e1,e2,e3), t = CA, x_self = R^T (pos - t).
// Storage: R[n*9 + i*3 + j] = e_{j}[i]
// ---------------------------------------------------------------------------
__global__ void k_frames(const float* __restrict__ pos, float* __restrict__ R,
                         float* __restrict__ T, float* __restrict__ Xs) {
    int n = blockIdx.x * blockDim.x + threadIdx.x;
    if (n >= NRES) return;
    const float* p = pos + n * NA * 3;
    float Nx = p[0], Ny = p[1], Nz = p[2];
    float ax = p[3], ay = p[4], az = p[5];
    float Cx = p[6], Cy = p[7], Cz = p[8];

    float e1x = Cx - ax, e1y = Cy - ay, e1z = Cz - az;
    float inv = rsqrtf(e1x * e1x + e1y * e1y + e1z * e1z + 1e-8f);
    e1x *= inv; e1y *= inv; e1z *= inv;

    float vx = Nx - ax, vy = Ny - ay, vz = Nz - az;
    float d = vx * e1x + vy * e1y + vz * e1z;
    float e2x = vx - d * e1x, e2y = vy - d * e1y, e2z = vz - d * e1z;
    inv = rsqrtf(e2x * e2x + e2y * e2y + e2z * e2z + 1e-8f);
    e2x *= inv; e2y *= inv; e2z *= inv;

    float e3x = e1y * e2z - e1z * e2y;
    float e3y = e1z * e2x - e1x * e2z;
    float e3z = e1x * e2y - e1y * e2x;

    float Rn[9] = { e1x, e2x, e3x,   e1y, e2y, e3y,   e1z, e2z, e3z };
#pragma unroll
    for (int i = 0; i < 9; ++i) R[n * 9 + i] = Rn[i];
    T[n * 3 + 0] = ax; T[n * 3 + 1] = ay; T[n * 3 + 2] = az;

#pragma unroll
    for (int a = 0; a < NA; ++a) {
        float px = p[a * 3 + 0] - ax, py = p[a * 3 + 1] - ay, pz = p[a * 3 + 2] - az;
#pragma unroll
        for (int j = 0; j < 3; ++j)
            Xs[n * 15 + a * 3 + j] = Rn[0 * 3 + j] * px + Rn[1 * 3 + j] * py + Rn[2 * 3 + j] * pz;
    }
}

// ---------------------------------------------------------------------------
// li = local @ W_li, lj = local @ W_lj via WMMA.
// ---------------------------------------------------------------------------
__global__ __launch_bounds__(256) void k_lilj(const float* __restrict__ local,
                                              const unsigned short* __restrict__ WtLiU,
                                              const unsigned short* __restrict__ WtLjU,
                                              float* __restrict__ Li, float* __restrict__ Lj) {
    const int tid = threadIdx.x, lane = tid & 31, wv = tid >> 5;
    const int row0 = blockIdx.x * 16;
    const int r = lane & 15, half = lane >> 4, base = half * 8;

#pragma unroll 1
    for (int m = 0; m < 2; ++m) {
        const _Float16* W = (const _Float16*)(m ? WtLjU : WtLiU);
        float* O = m ? Lj : Li;
        v8f acc = {};
#pragma unroll 1
        for (int kt = 0; kt < LOCALD / 32; ++kt) {
            const float* ap = &local[(row0 + r) * LOCALD + kt * 32 + base];
            v16h a;
#pragma unroll
            for (int i = 0; i < 8; ++i) { a[i] = (_Float16)ap[i]; a[i + 8] = (_Float16)ap[16 + i]; }
            v16h b = load_frag(&W[(wv * 16 + r) * LOCALD + kt * 32 + base]);
            acc = wmma16(a, b, acc);
        }
#pragma unroll
        for (int j = 0; j < 8; ++j)
            O[(row0 + j + 8 * half) * PAIRD + wv * 16 + r] = acc[j];
    }
}

// ---------------------------------------------------------------------------
// Main per-edge kernel: one block = 32 edges (global edge index g = blk*32+e,
// n = g/48, k = g%48).  LDS pool with lifetime-based overlays.
// ---------------------------------------------------------------------------
#define OFF_FEAT 0                       // 32*544*2 = 34816   (phases 1-2)
#define OFF_H    0                       // 32*256*2 = 16384   (phases 4-5, overlays FEAT)
#define OFF_P    34816                   // 32*128*4 = 16384   (phases 2-3)
#define OFF_PH   34816                   // 32*128*2 = 8192    (phases 3-4, overlays P)
#define OFF_PAIR 51200                   // 32*128*4 = 16384   (phases 0-3)
#define OFF_RED  67584                   // 512*8    = 4096
#define OFF_MS   71680                   // 32*8     = 256
#define OFF_NB   71936                   // 32*4     = 128
#define POOLSZ   72064

__global__ __launch_bounds__(256) void k_edges(
    const float* __restrict__ pair, const float* __restrict__ pos,
    const int* __restrict__ nbr, const unsigned char* __restrict__ msk,
    const float* __restrict__ lnS, const float* __restrict__ lnB,
    const float* __restrict__ b1, const float* __restrict__ b2,
    const float* __restrict__ R, const float* __restrict__ T,
    const float* __restrict__ Xs, const float* __restrict__ Li,
    const float* __restrict__ Lj,
    const unsigned short* __restrict__ WtAllU, const unsigned short* __restrict__ Wt1U,
    const unsigned short* __restrict__ Wt2U,
    float* __restrict__ out, float* __restrict__ outMask) {

    __shared__ alignas(16) unsigned char sPool[POOLSZ];
    _Float16* sFeat = (_Float16*)(sPool + OFF_FEAT);
    _Float16* sH    = (_Float16*)(sPool + OFF_H);
    float*    sP    = (float*)   (sPool + OFF_P);
    _Float16* sPh   = (_Float16*)(sPool + OFF_PH);
    float*    sPair = (float*)   (sPool + OFF_PAIR);
    float2*   sRed  = (float2*)  (sPool + OFF_RED);
    float2*   sMS   = (float2*)  (sPool + OFF_MS);
    int*      sNb   = (int*)     (sPool + OFF_NB);

    const _Float16* WtAll = (const _Float16*)WtAllU;
    const _Float16* Wt1   = (const _Float16*)Wt1U;
    const _Float16* Wt2   = (const _Float16*)Wt2U;

    const int tid  = threadIdx.x;
    const int lane = tid & 31;
    const int wv   = tid >> 5;
    const int g0   = blockIdx.x * EDG;

    // ---- Phase 0: async gather of pair[n,nb,:] rows into LDS ----
    {
#pragma unroll 1
        for (int i = 0; i < 4; ++i) {
            const int e  = wv * 4 + i;
            const int g  = g0 + e;
            const int ne = g / KNB, ke = g % KNB;
            const int nb = nbr[ne * KNB + ke];
            if (lane == 0) sNb[e] = nb;
            const float* src = pair + ((size_t)ne * NRES + nb) * PAIRD + lane * 4;
            float* dst = sPair + e * PAIRD + lane * 4;
            async_copy_b128(src, dst);
        }
    }

    // ---- Phase 1: geometric features (2 edges per thread) ----
#pragma unroll 1
    for (int t = 0; t < 2; ++t) {
        const int e = (tid >> 4) + t * 16;
        const int s = tid & 15;
        const int g = g0 + e;
        const int n = g / KNB, k = g % KNB;
        const int nb = nbr[n * KNB + k];

        float Rn[9], Tn[3];
#pragma unroll
        for (int i = 0; i < 9; ++i) Rn[i] = R[n * 9 + i];
#pragma unroll
        for (int i = 0; i < 3; ++i) Tn[i] = T[n * 3 + i];

        for (int pp = s; pp < NA * NA; pp += 16) {
            int a = pp / NA, b = pp % NA;
            float ox = pos[n * 15 + a * 3 + 0] - pos[nb * 15 + b * 3 + 0];
            float oy = pos[n * 15 + a * 3 + 1] - pos[nb * 15 + b * 3 + 1];
            float oz = pos[n * 15 + a * 3 + 2] - pos[nb * 15 + b * 3 + 2];
            float d = sqrtf(ox * ox + oy * oy + oz * oz + 1e-8f);
            // RBF: centers linspace(0,22,16) -> spacing 22/15; width 22/16
#pragma unroll
            for (int j = 0; j < 16; ++j) {
                float c = (22.0f / 15.0f) * (float)j;
                float z = (d - c) * (16.0f / 22.0f);
                sFeat[e * FPAD + pp * 16 + j] = (_Float16)__expf(-z * z);
            }
            float lx = Rn[0] * ox + Rn[3] * oy + Rn[6] * oz;
            float ly = Rn[1] * ox + Rn[4] * oy + Rn[7] * oz;
            float lz = Rn[2] * ox + Rn[5] * oy + Rn[8] * oz;
            float inv = rsqrtf(lx * lx + ly * ly + lz * lz + 1e-8f);
            sFeat[e * FPAD + 400 + pp * 3 + 0] = (_Float16)(lx * inv);
            sFeat[e * FPAD + 400 + pp * 3 + 1] = (_Float16)(ly * inv);
            sFeat[e * FPAD + 400 + pp * 3 + 2] = (_Float16)(lz * inv);
        }
        if (s == 1) {                           // R_rel = R[n]^T R[nb]
#pragma unroll
            for (int i = 0; i < 3; ++i)
#pragma unroll
                for (int j = 0; j < 3; ++j) {
                    float v = 0.0f;
#pragma unroll
                    for (int a = 0; a < 3; ++a) v += Rn[a * 3 + i] * R[nb * 9 + a * 3 + j];
                    sFeat[e * FPAD + 475 + i * 3 + j] = (_Float16)v;
                }
        }
        if (s == 2) {                           // vec = 0.1*[x_self | x_nb]
#pragma unroll
            for (int a = 0; a < NA; ++a) {
                float px = pos[nb * 15 + a * 3 + 0] - Tn[0];
                float py = pos[nb * 15 + a * 3 + 1] - Tn[1];
                float pz = pos[nb * 15 + a * 3 + 2] - Tn[2];
#pragma unroll
                for (int j = 0; j < 3; ++j) {
                    sFeat[e * FPAD + 484 + a * 6 + j] = (_Float16)(0.1f * Xs[n * 15 + a * 3 + j]);
                    float xnb = Rn[0 * 3 + j] * px + Rn[1 * 3 + j] * py + Rn[2 * 3 + j] * pz;
                    sFeat[e * FPAD + 484 + a * 6 + 3 + j] = (_Float16)(0.1f * xnb);
                }
            }
        }
        if (s == 3) {                           // K padding + pair mask
            for (int f = FREAL; f < FPAD; ++f) sFeat[e * FPAD + f] = (_Float16)0.0f;
            bool m = (msk[n] != 0) && (msk[nb] != 0) && (nb != -1);
            outMask[g] = m ? 1.0f : 0.0f;
        }
    }
    __syncthreads();

    // ---- Phase 2: feature GEMM (32x544)@(544x128); B shared by 2 A tiles ----
    {
        const int r = lane & 15, half = lane >> 4, base = half * 8;
        v8f acc0 = {}, acc1 = {};
#pragma unroll 1
        for (int kt = 0; kt < FPAD / 32; ++kt) {
            v16h b  = load_frag(&WtAll[(wv * 16 + r) * FPAD + kt * 32 + base]);
            v16h a0 = load_frag(&sFeat[r * FPAD + kt * 32 + base]);
            v16h a1 = load_frag(&sFeat[(16 + r) * FPAD + kt * 32 + base]);
            acc0 = wmma16(a0, b, acc0);
            acc1 = wmma16(a1, b, acc1);
        }
#pragma unroll
        for (int j = 0; j < 8; ++j) {
            sP[(j + 8 * half) * PAIRD + wv * 16 + r]        = acc0[j];
            sP[(16 + j + 8 * half) * PAIRD + wv * 16 + r]   = acc1[j];
        }
    }
    wait_async_all();
    __syncthreads();

    // ---- Phase 3: + pair gather + li/lj, LayerNorm ----
    {
        const int s = tid & 15;
        float v[2][8];
#pragma unroll 1
        for (int t = 0; t < 2; ++t) {
            const int e = (tid >> 4) + t * 16;
            const int g = g0 + e;
            const int n = g / KNB;
            const int nb = sNb[e];
            float sum = 0.0f, sq = 0.0f;
#pragma unroll
            for (int j = 0; j < 8; ++j) {
                int ch = s * 8 + j;
                float x = sP[e * PAIRD + ch] + sPair[e * PAIRD + ch]
                        + Li[n * PAIRD + ch] + Lj[nb * PAIRD + ch];
                v[t][j] = x; sum += x; sq += x * x;
            }
            sRed[e * 16 + s] = make_float2(sum, sq);
        }
        __syncthreads();
        if (s == 0) {
#pragma unroll 1
            for (int t = 0; t < 2; ++t) {
                const int e = (tid >> 4) + t * 16;
                float S = 0.0f, Q = 0.0f;
                for (int i = 0; i < 16; ++i) { float2 t2 = sRed[e * 16 + i]; S += t2.x; Q += t2.y; }
                float mu  = S * (1.0f / PAIRD);
                float var = Q * (1.0f / PAIRD) - mu * mu;
                sMS[e] = make_float2(mu, rsqrtf(var + 1e-5f));
            }
        }
        __syncthreads();
#pragma unroll 1
        for (int t = 0; t < 2; ++t) {
            const int e = (tid >> 4) + t * 16;
            float2 ms = sMS[e];
#pragma unroll
            for (int j = 0; j < 8; ++j) {
                int ch = s * 8 + j;
                float x = (v[t][j] - ms.x) * ms.y * lnS[ch] + lnB[ch];
                sPh[e * PAIRD + ch] = (_Float16)x;
            }
        }
    }
    __syncthreads();

    // ---- Phase 4: MLP layer 1 (32x128)@(128x256) + GELU ----
    {
        const int r = lane & 15, half = lane >> 4, base = half * 8;
#pragma unroll 1
        for (int sub = 0; sub < 2; ++sub) {
            const int nt = wv + sub * 8;
            v8f acc0 = {}, acc1 = {};
#pragma unroll
            for (int kt = 0; kt < 4; ++kt) {
                v16h b  = load_frag(&Wt1[(nt * 16 + r) * PAIRD + kt * 32 + base]);
                v16h a0 = load_frag(&sPh[r * PAIRD + kt * 32 + base]);
                v16h a1 = load_frag(&sPh[(16 + r) * PAIRD + kt * 32 + base]);
                acc0 = wmma16(a0, b, acc0);
                acc1 = wmma16(a1, b, acc1);
            }
            const int col = nt * 16 + r;
            const float bb = b1[col];
#pragma unroll
            for (int j = 0; j < 8; ++j) {
                float x = acc0[j] + bb;
                float gx = 0.5f * x * (1.0f + tanhf(0.7978845608028654f * (x + 0.044715f * x * x * x)));
                sH[(j + 8 * half) * 256 + col] = (_Float16)gx;
                x = acc1[j] + bb;
                gx = 0.5f * x * (1.0f + tanhf(0.7978845608028654f * (x + 0.044715f * x * x * x)));
                sH[(16 + j + 8 * half) * 256 + col] = (_Float16)gx;
            }
        }
    }
    __syncthreads();

    // ---- Phase 5: MLP layer 2 (32x256)@(256x128) + store ----
    {
        const int r = lane & 15, half = lane >> 4, base = half * 8;
        v8f acc0 = {}, acc1 = {};
#pragma unroll 1
        for (int kt = 0; kt < 8; ++kt) {
            v16h b  = load_frag(&Wt2[(wv * 16 + r) * 256 + kt * 32 + base]);
            v16h a0 = load_frag(&sH[r * 256 + kt * 32 + base]);
            v16h a1 = load_frag(&sH[(16 + r) * 256 + kt * 32 + base]);
            acc0 = wmma16(a0, b, acc0);
            acc1 = wmma16(a1, b, acc1);
        }
        const int col = wv * 16 + r;
        const float bb = b2[col];
#pragma unroll
        for (int j = 0; j < 8; ++j) {
            out[((long long)(g0 + j + 8 * half)) * PAIRD + col]      = acc0[j] + bb;
            out[((long long)(g0 + 16 + j + 8 * half)) * PAIRD + col] = acc1[j] + bb;
        }
    }
}

// ---------------------------------------------------------------------------
extern "C" void kernel_launch(void* const* d_in, const int* in_sizes, int n_in,
                              void* d_out, int out_size, void* d_ws, size_t ws_size,
                              hipStream_t stream) {
    (void)in_sizes; (void)n_in; (void)out_size; (void)ws_size;
    const float* local  = (const float*)d_in[0];
    const float* pos    = (const float*)d_in[1];
    const float* pair   = (const float*)d_in[2];
    const int*   nbr    = (const int*)d_in[3];
    const unsigned char* msk = (const unsigned char*)d_in[4];
    const float* W_li   = (const float*)d_in[5];
    const float* W_lj   = (const float*)d_in[6];
    const float* W_dist = (const float*)d_in[7];
    const float* W_dir  = (const float*)d_in[8];
    const float* W_rot  = (const float*)d_in[9];
    const float* W_vec  = (const float*)d_in[10];
    const float* lnS    = (const float*)d_in[11];
    const float* lnB    = (const float*)d_in[12];
    const float* w1     = (const float*)d_in[13];
    const float* b1     = (const float*)d_in[14];
    const float* w2     = (const float*)d_in[15];
    const float* b2     = (const float*)d_in[16];

    char* ws = (char*)d_ws;
    size_t off = 0;
    auto alloc = [&](size_t bytes) -> void* {
        void* p = ws + off;
        off += (bytes + 255) & ~(size_t)255;
        return p;
    };
    float* Rf = (float*)alloc(NRES * 9 * sizeof(float));
    float* Tf = (float*)alloc(NRES * 3 * sizeof(float));
    float* Xs = (float*)alloc(NRES * 15 * sizeof(float));
    float* Li = (float*)alloc((size_t)NRES * PAIRD * sizeof(float));
    float* Lj = (float*)alloc((size_t)NRES * PAIRD * sizeof(float));
    unsigned short* WtAll = (unsigned short*)alloc((size_t)PAIRD * FPAD * 2);
    unsigned short* Wt1   = (unsigned short*)alloc((size_t)256 * 128 * 2);
    unsigned short* Wt2   = (unsigned short*)alloc((size_t)128 * 256 * 2);
    unsigned short* WtLi  = (unsigned short*)alloc((size_t)128 * 512 * 2);
    unsigned short* WtLj  = (unsigned short*)alloc((size_t)128 * 512 * 2);

    k_prep_w<<<1040, 256, 0, stream>>>(W_dist, W_dir, W_rot, W_vec, w1, w2, W_li, W_lj,
                                       WtAll, Wt1, Wt2, WtLi, WtLj);
    k_frames<<<4, 256, 0, stream>>>(pos, Rf, Tf, Xs);
    k_lilj<<<NRES / 16, 256, 0, stream>>>(local, WtLi, WtLj, Li, Lj);

    float* out = (float*)d_out;
    float* outMask = out + (size_t)NRES * KNB * PAIRD;
    k_edges<<<(NRES * KNB) / EDG, 256, 0, stream>>>(pair, pos, nbr, msk, lnS, lnB, b1, b2,
                                                    Rf, Tf, Xs, Li, Lj, WtAll, Wt1, Wt2,
                                                    out, outMask);
}
